// CrossLevelAttention_54039278518785
// MI455X (gfx1250) — compile-verified, matrix operand order checked by reading, not run on hardware
//
#include <hip/hip_runtime.h>

#define D   1024
#define TM  32                 // batch rows per workgroup
#define NT  (D / 16)           // 64 N-tiles
#define N1  (D * D)            // 1<<20

typedef __attribute__((ext_vector_type(16))) __bf16       v16bf;
typedef __attribute__((ext_vector_type(8)))  float        v8f;
typedef __attribute__((ext_vector_type(4)))  unsigned int v4u;

// round-to-nearest-even f32 -> bf16
static __device__ __forceinline__ __bf16 f2bf(float f) {
    unsigned u = __builtin_bit_cast(unsigned, f);
    u += 0x7FFFu + ((u >> 16) & 1u);
    unsigned short h = (unsigned short)(u >> 16);
    return __builtin_bit_cast(__bf16, h);
}

// Load one 16x32 bf16 WMMA operand fragment for this lane.
// Per CDNA5 ISA: lanes 0-15 hold K=0..7 (v[0..7]) and K=16..23 (v[8..15]);
// lanes 16-31 hold K=8..15 and K=24..31.  p already includes the +8 half shift.
static __device__ __forceinline__ v16bf load_frag(const __bf16* p) {
    v16bf r;
    v4u* rp = (v4u*)&r;
    rp[0] = *(const v4u*)(p);        // K = ka .. ka+7
    rp[1] = *(const v4u*)(p + 16);   // K = ka+16 .. ka+23
    return r;
}

// Warm a TM x D f32 tile (128 KB = 1024 cachelines of 128 B) into cache during
// the following GEMM: 256 threads x 4 lines each.  global_prefetch_b8 — no
// counters, no VGPR results, pure latency hiding for the residual read in LN.
static __device__ __forceinline__ void prefetch_tile(const float* p, int tid) {
    const char* c = (const char*)p;
#pragma unroll
    for (int i = 0; i < 4; ++i)
        __builtin_prefetch(c + ((size_t)tid * 4 + i) * 128, 0, 3);
}

// Y(32xD) = X(32xD, LDS bf16) @ W^T(DxD, row-major bf16, row n = output col n) + bias
// Each wave covers N-tiles {wave, wave+8, ...}; two M-tiles (rows 0-15, 16-31).
template <bool F32OUT>
static __device__ __forceinline__ void gemm_tile(const __bf16* __restrict__ A,
                                                 const __bf16* __restrict__ W,
                                                 const float*  __restrict__ bias,
                                                 __bf16* Vout, float* Fout,
                                                 int lane, int wave) {
    const int l16  = lane & 15;
    const int half = lane >> 4;
    const __bf16* arow0 = A + (size_t)l16 * D;
    const __bf16* arow1 = A + (size_t)(l16 + 16) * D;
    for (int nt = wave; nt < NT; nt += 8) {
        const int ncol = nt * 16 + l16;
        const __bf16* wrow = W + (size_t)ncol * D;   // B lane n == W row n (no transpose)
        v8f c0 = {0.f, 0.f, 0.f, 0.f, 0.f, 0.f, 0.f, 0.f};
        v8f c1 = c0;
#pragma unroll 4
        for (int k0 = 0; k0 < D; k0 += 32) {
            const int ka = k0 + half * 8;
            v16bf a0 = load_frag(arow0 + ka);
            v16bf a1 = load_frag(arow1 + ka);
            v16bf bf = load_frag(wrow + ka);
            c0 = __builtin_amdgcn_wmma_f32_16x16x32_bf16(false, a0, false, bf,
                                                         (short)0, c0, false, false);
            c1 = __builtin_amdgcn_wmma_f32_16x16x32_bf16(false, a1, false, bf,
                                                         (short)0, c1, false, false);
        }
        const float bn = bias[ncol];
#pragma unroll
        for (int j = 0; j < 8; ++j) {             // C/D layout: VGPR j -> M=j (+8 for hi lanes)
            const int m0 = j + half * 8;
            const int m1 = m0 + 16;
            const float y0 = c0[j] + bn;
            const float y1 = c1[j] + bn;
            if (F32OUT) {
                Fout[(size_t)m0 * D + ncol] = y0;
                Fout[(size_t)m1 * D + ncol] = y1;
            } else {
                Vout[(size_t)m0 * D + ncol] = f2bf(y0);
                Vout[(size_t)m1 * D + ncol] = f2bf(y1);
            }
        }
    }
}

// residual add (+ optional LayerNorm), write f32 to global, bf16 copy to Xbf for next GEMM.
// 8 threads per row, 128 cols each.
static __device__ __forceinline__ void resid_ln(float* Yf, const float* __restrict__ resid,
                                                const float* __restrict__ g,
                                                const float* __restrict__ b,
                                                float* __restrict__ outg, __bf16* Xbf,
                                                float* red, int tid, bool do_ln) {
    const int row = tid >> 3;
    const int sub = tid & 7;
    float4*       Y4 = (float4*)(Yf + (size_t)row * D + sub * 128);
    const float4* R4 = (const float4*)(resid + (size_t)row * D + sub * 128);
    float4*       O4 = (float4*)(outg + (size_t)row * D + sub * 128);
    float s = 0.f, s2 = 0.f;
#pragma unroll 8
    for (int i = 0; i < 32; ++i) {
        float4 y = Y4[i];
        const float4 r = R4[i];
        y.x += r.x; y.y += r.y; y.z += r.z; y.w += r.w;
        Y4[i] = y;
        s  += y.x + y.y + y.z + y.w;
        s2 += y.x * y.x + y.y * y.y + y.z * y.z + y.w * y.w;
    }
    if (!do_ln) {
#pragma unroll 8
        for (int i = 0; i < 32; ++i) O4[i] = Y4[i];
        return;
    }
    red[(row * 8 + sub) * 2 + 0] = s;
    red[(row * 8 + sub) * 2 + 1] = s2;
    __syncthreads();
    float m = 0.f, v = 0.f;
#pragma unroll
    for (int i = 0; i < 8; ++i) {
        m += red[(row * 8 + i) * 2 + 0];
        v += red[(row * 8 + i) * 2 + 1];
    }
    m *= (1.f / D);
    v  = v * (1.f / D) - m * m;
    const float rstd = rsqrtf(v + 1e-5f);
#pragma unroll 4
    for (int i = 0; i < 32; ++i) {
        const int c = sub * 128 + i * 4;
        float4 y = Y4[i];
        y.x = (y.x - m) * rstd * g[c + 0] + b[c + 0];
        y.y = (y.y - m) * rstd * g[c + 1] + b[c + 1];
        y.z = (y.z - m) * rstd * g[c + 2] + b[c + 2];
        y.w = (y.w - m) * rstd * g[c + 3] + b[c + 3];
        O4[i] = y;
        __bf16* xb = Xbf + (size_t)row * D + c;
        xb[0] = f2bf(y.x); xb[1] = f2bf(y.y); xb[2] = f2bf(y.z); xb[3] = f2bf(y.w);
    }
}

__global__ __launch_bounds__(256) void fused_chain_kernel(
    const float* __restrict__ local_f, const float* __restrict__ tact_f,
    const float* __restrict__ strat_f,
    const __bf16* __restrict__ Wall,           // [0..3]=Wv blk0..3, [4..7]=Wout blk0..3
    const float* __restrict__ bv,              // 4*D
    const float* __restrict__ bo,              // 4*D
    const float* __restrict__ ln1g, const float* __restrict__ ln1b,
    const float* __restrict__ ln2g, const float* __restrict__ ln2b,
    const float* __restrict__ ln3g, const float* __restrict__ ln3b,
    float* __restrict__ outLoc, float* __restrict__ outTac, float* __restrict__ outStrat) {
    extern __shared__ char smem[];
    __bf16* Xbf = (__bf16*)smem;                      //  64 KB: current GEMM input
    __bf16* Vbf = (__bf16*)(smem + TM * D * 2);       //  64 KB: intermediate v
    float*  Yf  = (float*)(smem + TM * D * 4);        // 128 KB: pre-LN f32
    float*  red = (float*)(smem + (size_t)TM * D * 8);//   2 KB: LN reduction scratch

    const int tid  = threadIdx.x;
    const int lane = tid & 31;
    const int wave = tid >> 5;
    const size_t base = (size_t)blockIdx.x * TM * D;

    // stage local tile -> Xbf (bf16)
    {
        const float4* src = (const float4*)(local_f + base);
        for (int i = tid; i < TM * D / 4; i += 256) {
            const float4 v = src[i];
            __bf16* x = Xbf + (size_t)i * 4;
            x[0] = f2bf(v.x); x[1] = f2bf(v.y); x[2] = f2bf(v.z); x[3] = f2bf(v.w);
        }
    }
    __syncthreads();

    // block 1 (l2t): tac = LN1(tactical + (local@Wv0^T+bv0)@Wo0^T+bo0)
    gemm_tile<false>(Xbf, Wall + (size_t)0 * N1, bv + 0 * D, Vbf, nullptr, lane, wave);
    __syncthreads();
    prefetch_tile(tact_f + base, tid);                       // warm residual during GEMM2
    gemm_tile<true>(Vbf, Wall + (size_t)4 * N1, bo + 0 * D, nullptr, Yf, lane, wave);
    __syncthreads();
    resid_ln(Yf, tact_f + base, ln1g, ln1b, outTac + base, Xbf, red, tid, true);
    __syncthreads();

    // block 2 (t2s): strat = LN2(strategic + ...)
    gemm_tile<false>(Xbf, Wall + (size_t)1 * N1, bv + 1 * D, Vbf, nullptr, lane, wave);
    __syncthreads();
    prefetch_tile(strat_f + base, tid);
    gemm_tile<true>(Vbf, Wall + (size_t)5 * N1, bo + 1 * D, nullptr, Yf, lane, wave);
    __syncthreads();
    resid_ln(Yf, strat_f + base, ln2g, ln2b, outStrat + base, Xbf, red, tid, true);
    __syncthreads();

    // block 3 (s2t): tac = LN3(tac + ...); residual re-read from outTac (written in block 1)
    gemm_tile<false>(Xbf, Wall + (size_t)2 * N1, bv + 2 * D, Vbf, nullptr, lane, wave);
    __syncthreads();
    prefetch_tile(outTac + base, tid);
    gemm_tile<true>(Vbf, Wall + (size_t)6 * N1, bo + 2 * D, nullptr, Yf, lane, wave);
    __syncthreads();
    resid_ln(Yf, outTac + base, ln3g, ln3b, outTac + base, Xbf, red, tid, true);
    __syncthreads();

    // block 4 (t2l): loc = local + ... (no LN)
    gemm_tile<false>(Xbf, Wall + (size_t)3 * N1, bv + 3 * D, Vbf, nullptr, lane, wave);
    __syncthreads();
    prefetch_tile(local_f + base, tid);
    gemm_tile<true>(Vbf, Wall + (size_t)7 * N1, bo + 3 * D, nullptr, Yf, lane, wave);
    __syncthreads();
    resid_ln(Yf, local_f + base, ln1g, ln1b, outLoc + base, nullptr, red, tid, false);
}

// Convert the V-slice of each w_in (rows 2D..3D) and each w_out to bf16 in d_ws;
// copy V-slice biases and out biases to f32 arrays.
__global__ void prep_weights_kernel(
    const float* wi0, const float* wi1, const float* wi2, const float* wi3,
    const float* bi0, const float* bi1, const float* bi2, const float* bi3,
    const float* wo0, const float* wo1, const float* wo2, const float* wo3,
    const float* bo0, const float* bo1, const float* bo2, const float* bo3,
    __bf16* Wall, float* bv, float* bo) {
    const float* wis[4] = {wi0, wi1, wi2, wi3};
    const float* wos[4] = {wo0, wo1, wo2, wo3};
    const float* bis[4] = {bi0, bi1, bi2, bi3};
    const float* bos[4] = {bo0, bo1, bo2, bo3};
    const size_t tid    = (size_t)blockIdx.x * blockDim.x + threadIdx.x;
    const size_t stride = (size_t)gridDim.x * blockDim.x;
    for (size_t i = tid; i < (size_t)8 * N1; i += stride) {
        const int    blk = (int)(i >> 20) & 3;
        const size_t e   = i & (size_t)(N1 - 1);
        if (i < (size_t)4 * N1)
            Wall[i] = f2bf(wis[blk][(size_t)2 * N1 + e]);   // Wv = w_in[2D:3D]
        else
            Wall[i] = f2bf(wos[blk][e]);
    }
    if (tid < 8192) {
        const int blk = (int)(tid >> 10) & 3;
        const int d   = (int)tid & (D - 1);
        if (tid < 4096) bv[tid] = bis[blk][2 * D + d];      // bv = b_in[2D:3D]
        else            bo[tid - 4096] = bos[blk][d];
    }
}

extern "C" void kernel_launch(void* const* d_in, const int* in_sizes, int n_in,
                              void* d_out, int out_size, void* d_ws, size_t ws_size,
                              hipStream_t stream) {
    const float* local_f = (const float*)d_in[0];
    const float* tact_f  = (const float*)d_in[1];
    const float* strat_f = (const float*)d_in[2];
    const float* w_in[4]; const float* b_in[4]; const float* w_out[4]; const float* b_out[4];
    for (int i = 0; i < 4; ++i) {
        w_in[i]  = (const float*)d_in[3 + i * 4 + 0];
        b_in[i]  = (const float*)d_in[3 + i * 4 + 1];
        w_out[i] = (const float*)d_in[3 + i * 4 + 2];
        b_out[i] = (const float*)d_in[3 + i * 4 + 3];
    }
    const float* ln1g = (const float*)d_in[19];
    const float* ln1b = (const float*)d_in[20];
    const float* ln2g = (const float*)d_in[21];
    const float* ln2b = (const float*)d_in[22];
    const float* ln3g = (const float*)d_in[23];
    const float* ln3b = (const float*)d_in[24];

    __bf16* Wall = (__bf16*)d_ws;                                     // 16 MB bf16 weights
    float*  bv   = (float*)((char*)d_ws + (size_t)8 * N1 * 2);        // 16 KB
    float*  bo   = bv + 4 * D;                                        // 16 KB

    const int rows = in_sizes[0] / D;   // 65536
    float* out      = (float*)d_out;
    float* outLoc   = out;
    float* outTac   = out + (size_t)rows * D;
    float* outStrat = out + (size_t)2 * rows * D;

    prep_weights_kernel<<<4096, 256, 0, stream>>>(
        w_in[0], w_in[1], w_in[2], w_in[3], b_in[0], b_in[1], b_in[2], b_in[3],
        w_out[0], w_out[1], w_out[2], w_out[3], b_out[0], b_out[1], b_out[2], b_out[3],
        Wall, bv, bo);

    const size_t smem = (size_t)TM * D * 8 + (size_t)TM * 8 * 2 * 4;  // 264,192 B (<320 KB/WGP)
    fused_chain_kernel<<<rows / TM, 256, smem, stream>>>(
        local_f, tact_f, strat_f, Wall, bv, bo,
        ln1g, ln1b, ln2g, ln2b, ln3g, ln3b,
        outLoc, outTac, outStrat);
}
